// GAT_53180285059632
// MI455X (gfx1250) — matrix operand order repaired and checked by reading, not compiled, not run on hardware
//
#include <hip/hip_runtime.h>
#include <hip/hip_bf16.h>

// ---------------- problem constants (match reference) ----------------
#define NN   100000
#define EE   400000
#define EP   500000          // E + N self loops
#define LL   2
#define HH   4
#define CC   64
#define FF   64
#define HC   256
#define GG   512
#define DD   256
#define KK   2
#define NEG_SLOPE 0.2f

typedef __attribute__((ext_vector_type(16))) _Float16 v16h;
typedef __attribute__((ext_vector_type(8)))  float    v8f;
typedef __attribute__((ext_vector_type(4)))  unsigned u32x4;
typedef __attribute__((ext_vector_type(8)))  int      i32x8;
typedef __attribute__((ext_vector_type(4)))  int      i32x4;

// ---------------- ordered-uint mapping for float atomic max ----------------
__device__ __forceinline__ unsigned f2ord(float f) {
    unsigned u = __float_as_uint(f);
    return (u & 0x80000000u) ? ~u : (u | 0x80000000u);
}
__device__ __forceinline__ float ord2f(unsigned u) {
    u = (u & 0x80000000u) ? (u & 0x7fffffffu) : ~u;
    return __uint_as_float(u);
}
#define ORD_NEG_INF 0x007FFFFFu   // f2ord(-inf)

// ---------------- WMMA fragment swizzle helpers (ISA 7.12.2) ----------------
// A 16x32 f16: lane = hi*16 + m (m = row%16), element e = 2*v + h where
//   kk = k%32 :  hi = (kk>>3)&1 ; v = ((kk&7)>>1) + ((kk>>4)<<2) ; h = kk&1
__device__ __forceinline__ int a_frag_slot(int kk, int* lane_hi) {
    *lane_hi = (kk >> 3) & 1;
    int v = ((kk & 7) >> 1) + ((kk >> 4) << 2);
    return 2 * v + (kk & 1);
}
// B 32x16 f16: lane = hi*16 + n%16 ; kk = k%32 : hi = (kk>>3)&1 ; v = ((kk&7)>>1) + ((kk>>4)<<2) ; h = kk&1
__device__ __forceinline__ int b_frag_slot(int kk, int* lane_hi) {
    *lane_hi = (kk >> 3) & 1;
    int v = ((kk & 7) >> 1) + ((kk >> 4) << 2);
    return 2 * v + (kk & 1);
}

// ---------------- Tensor Data Mover: 2-D f32 tile -> LDS ----------------
// D# per ISA 8.3/8.4: group0 = {flags, lds_addr, global_addr, type=2},
// group1 = {data_size, tensor dims, tile dims, dim0 stride}. Groups 2/3 zero (<=2D).
// This toolchain exposes the 6-arg builtin: (g0, g1, g2, g3, g4, cpol).
__device__ __forceinline__ void tdm_load_2d_f32(unsigned lds_off,
                                                const float* gptr,
                                                unsigned rows, unsigned cols,
                                                unsigned stride_elems) {
    unsigned long long ga = (unsigned long long)(size_t)gptr;
    u32x4 g0;
    g0[0] = 1u;                                           // count=1, user mode
    g0[1] = lds_off;                                      // lds_addr (bytes)
    g0[2] = (unsigned)(ga & 0xffffffffu);                 // global_addr[31:0]
    g0[3] = (unsigned)((ga >> 32) & 0x01ffffffu) | (2u << 30); // addr[56:32] | type=2
    i32x8 g1;
    g1[0] = (int)(2u << 16);                              // data_size=2 (4B), wg_mask=0
    g1[1] = (int)((cols & 0xffffu) << 16);                // tensor_dim0[15:0]
    g1[2] = (int)(((cols >> 16) & 0xffffu) | ((rows & 0xffffu) << 16)); // dim0 hi | dim1 lo
    g1[3] = (int)(((rows >> 16) & 0xffffu) | ((cols & 0xffffu) << 16)); // dim1 hi | tile_dim0
    g1[4] = (int)(rows & 0xffffu);                        // tile_dim1 (tile_dim2 = 0)
    g1[5] = (int)stride_elems;                            // tensor_dim0_stride[31:0]
    g1[6] = 0;
    g1[7] = 0;
    i32x4 z4 = {0, 0, 0, 0};
    i32x8 z8 = {0, 0, 0, 0, 0, 0, 0, 0};
    __builtin_amdgcn_tensor_load_to_lds(g0, g1, z4, z4, z8, 0);
}

// ---------------- GEMM1: xl[N,256] = x[N,64] @ Wg[64,256] ----------------
// block = 256 thr (8 waves); tile = 32 rows x 256 cols; wave = 16 rows x 64 cols
// Dynamic LDS: [0,64K) Wg f32 staging (TDM) | [64K,96K) B frags | [96K,100K) A frags
#define G1_LDS (65536 + 32768 + 4096)
__global__ void gat_gemm1_kernel(const float* __restrict__ x,
                                 const float* __restrict__ W,
                                 float* __restrict__ xl) {
    extern __shared__ char smem[];
    float*    Wstage  = (float*)smem;                         // 64*256 f32
    _Float16* Bs_frag = (_Float16*)(smem + 65536);            // [2 ks][16 nt][32][16]
    _Float16* As_frag = (_Float16*)(smem + 65536 + 32768);    // [2 rt][2 ks][32][16]
    int tid = threadIdx.x;
    int block_row = blockIdx.x * 32;

    // async DMA of the weight matrix into LDS (issued by wave 0 only)
    if (tid < 32) {
        tdm_load_2d_f32((unsigned)(size_t)Wstage, W, 64u, 256u, 256u);
        __builtin_amdgcn_s_wait_tensorcnt(0);
    }

    // stage A (global f32 -> f16, pre-swizzled to fragment layout)
    for (int i = tid; i < 32 * 64; i += 256) {
        int r = i >> 6, c = i & 63;
        int rt = r >> 4, m = r & 15, ks = c >> 5, kk = c & 31;
        int hi, e = a_frag_slot(kk, &hi);
        As_frag[(((rt * 2 + ks) * 32) + hi * 16 + m) * 16 + e] =
            (_Float16)x[(size_t)(block_row + r) * 64 + c];
    }
    __syncthreads();   // Wstage ready for all waves

    // stage B (LDS f32 -> f16, pre-swizzled)
    for (int i = tid; i < 64 * 256; i += 256) {
        int k = i >> 8, n = i & 255;
        int ks = k >> 5, kk = k & 31;
        int hi, e = b_frag_slot(kk, &hi);
        int nt = n >> 4;
        Bs_frag[(((ks * 16 + nt) * 32) + hi * 16 + (n & 15)) * 16 + e] =
            (_Float16)Wstage[i];
    }
    __syncthreads();

    int wid = tid >> 5, lane = tid & 31;
    int rt = wid >> 2;               // 0 / 1  (16-row tile)
    int cg = wid & 3;                // 0..3   (64-col group)
    v8f acc[4] = {};
#pragma unroll
    for (int ks = 0; ks < 2; ++ks) {
        v16h a = *(const v16h*)(As_frag + (((rt * 2 + ks) * 32) + lane) * 16);
#pragma unroll
        for (int j = 0; j < 4; ++j) {
            int nt = cg * 4 + j;
            v16h b = *(const v16h*)(Bs_frag + (((ks * 16 + nt) * 32) + lane) * 16);
            acc[j] = __builtin_amdgcn_wmma_f32_16x16x32_f16(
                false, a, false, b, (short)0, acc[j], false, false);
        }
    }
    int n0 = lane & 15, mb = (lane >> 4) * 8;
#pragma unroll
    for (int j = 0; j < 4; ++j)
#pragma unroll
        for (int v = 0; v < 8; ++v)
            xl[(size_t)(block_row + rt * 16 + mb + v) * 256 +
               (cg * 4 + j) * 16 + n0] = acc[j][v];
}

// ---- GEMM2: xout[N,64] = relu( relu(acc_in+bg)[N,256] @ Wl^T[256,64] + bl ) ----
// Dynamic LDS: [0,64K) Wl f32 staging (TDM) | [64K,96K) B frags | [96K,112K) A frags
#define G2_LDS (65536 + 32768 + 16384)
__global__ void gat_gemm2_kernel(const float* __restrict__ acc_in,
                                 const float* __restrict__ bg,
                                 const float* __restrict__ Wl,
                                 const float* __restrict__ bl,
                                 float* __restrict__ xout) {
    extern __shared__ char smem[];
    float*    Wstage  = (float*)smem;                         // Wl: [64 rows][256]
    _Float16* Bs_frag = (_Float16*)(smem + 65536);            // [8 ks][4 nt][32][16]
    _Float16* As_frag = (_Float16*)(smem + 65536 + 32768);    // [2 rt][8 ks][32][16]
    int tid = threadIdx.x;
    int block_row = blockIdx.x * 32;

    if (tid < 32) {
        tdm_load_2d_f32((unsigned)(size_t)Wstage, Wl, 64u, 256u, 256u);
        __builtin_amdgcn_s_wait_tensorcnt(0);
    }

    // stage A = relu(acc_in + bg), pre-swizzled
    for (int i = tid; i < 32 * 256; i += 256) {
        int r = i >> 8, c = i & 255;
        float v = acc_in[(size_t)(block_row + r) * 256 + c] + bg[c];
        v = v > 0.f ? v : 0.f;
        int rt = r >> 4, m = r & 15, ks = c >> 5, kk = c & 31;
        int hi, e = a_frag_slot(kk, &hi);
        As_frag[(((rt * 8 + ks) * 32) + hi * 16 + m) * 16 + e] = (_Float16)v;
    }
    __syncthreads();   // Wstage ready

    // stage B[k][n] = Wl[n][k]  (transpose + convert from LDS staging)
    for (int i = tid; i < 64 * 256; i += 256) {
        int n = i >> 8, k = i & 255;     // sequential reads of Wstage
        int ks = k >> 5, kk = k & 31;
        int hi, e = b_frag_slot(kk, &hi);
        int nt = n >> 4;
        Bs_frag[(((ks * 4 + nt) * 32) + hi * 16 + (n & 15)) * 16 + e] =
            (_Float16)Wstage[i];
    }
    __syncthreads();

    int wid = tid >> 5, lane = tid & 31;
    int rt = wid >> 2;               // 0 / 1
    int nt = wid & 3;                // 16-col tile
    v8f acc = {};
#pragma unroll
    for (int ks = 0; ks < 8; ++ks) {
        v16h a = *(const v16h*)(As_frag + (((rt * 8 + ks) * 32) + lane) * 16);
        v16h b = *(const v16h*)(Bs_frag + (((ks * 4 + nt) * 32) + lane) * 16);
        acc = __builtin_amdgcn_wmma_f32_16x16x32_f16(
            false, a, false, b, (short)0, acc, false, false);
    }
    int n0 = lane & 15, mb = (lane >> 4) * 8;
#pragma unroll
    for (int v = 0; v < 8; ++v) {
        int row = block_row + rt * 16 + mb + v, col = nt * 16 + n0;
        float r = acc[v] + bl[col];
        xout[(size_t)row * 64 + col] = r > 0.f ? r : 0.f;
    }
}

// ---------------- per-node attention dot products ----------------
__global__ void gat_alpha_kernel(const float* __restrict__ xl,
                                 const float* __restrict__ a_s,
                                 const float* __restrict__ a_d,
                                 float* __restrict__ als,
                                 float* __restrict__ ald) {
    int i = blockIdx.x * blockDim.x + threadIdx.x;   // over N*H
    if (i >= NN * HH) return;
    int n = i >> 2, h = i & 3;
    const float4* xp = (const float4*)(xl + (size_t)n * 256 + h * 64);
    const float4* sp = (const float4*)(a_s + h * 64);
    const float4* dp = (const float4*)(a_d + h * 64);
    float ss = 0.f, dd = 0.f;
#pragma unroll
    for (int j = 0; j < 16; ++j) {
        float4 xv = xp[j], sv = sp[j], dv = dp[j];
        ss += xv.x * sv.x + xv.y * sv.y + xv.z * sv.z + xv.w * sv.w;
        dd += xv.x * dv.x + xv.y * dv.y + xv.z * dv.z + xv.w * dv.w;
    }
    als[i] = ss;
    ald[i] = dd;
}

// ---------------- init (per layer) ----------------
__global__ void gat_init_layer_kernel(float* __restrict__ outacc,
                                      unsigned* __restrict__ amax,
                                      float* __restrict__ denom) {
    int i = blockIdx.x * blockDim.x + threadIdx.x;
    if (i < NN * HC) outacc[i] = 0.f;
    if (i < NN * HH) { amax[i] = 0u; denom[i] = 0.f; }
}

// ---------------- edge pass 1: leaky-relu alpha + segment max ----------------
__global__ void gat_edge_max_kernel(const int* __restrict__ srcv,
                                    const int* __restrict__ dstv,
                                    const float* __restrict__ als,
                                    const float* __restrict__ ald,
                                    unsigned* __restrict__ amax,
                                    float* __restrict__ alpha_ws) {
    int i = blockIdx.x * blockDim.x + threadIdx.x;   // over EP*H
    if (i >= EP * HH) return;
    int e = i >> 2, h = i & 3;
    int s = (e < EE) ? srcv[e] : (e - EE);
    int d = (e < EE) ? dstv[e] : (e - EE);
    float a = als[s * HH + h] + ald[d * HH + h];
    a = a > 0.f ? a : NEG_SLOPE * a;
    alpha_ws[i] = a;
    atomicMax(&amax[d * HH + h], f2ord(a));
}

// ---------------- edge pass 2: exp + segment sum ----------------
__global__ void gat_edge_exp_kernel(const int* __restrict__ dstv,
                                    const unsigned* __restrict__ amax,
                                    float* __restrict__ alpha_ws,
                                    float* __restrict__ denom) {
    int i = blockIdx.x * blockDim.x + threadIdx.x;
    if (i >= EP * HH) return;
    int e = i >> 2, h = i & 3;
    int d = (e < EE) ? dstv[e] : (e - EE);
    float m = ord2f(amax[d * HH + h]);
    float ex = __expf(alpha_ws[i] - m);
    alpha_ws[i] = ex;
    atomicAdd(&denom[d * HH + h], ex);
}

// ---------------- edge pass 3: weighted scatter (wave per edge) ----------------
__global__ void gat_scatter_kernel(const int* __restrict__ srcv,
                                   const int* __restrict__ dstv,
                                   const float* __restrict__ xl,
                                   const float* __restrict__ exw,
                                   const float* __restrict__ denom,
                                   float* __restrict__ outacc) {
    int gtid = blockIdx.x * 256 + threadIdx.x;
    int e = gtid >> 5;
    if (e >= EP) return;
    int lane = gtid & 31;
    int s = (e < EE) ? srcv[e] : (e - EE);
    int d = (e < EE) ? dstv[e] : (e - EE);
#pragma unroll
    for (int half = 0; half < 2; ++half) {
        int col = half * 128 + lane * 4;      // 2 x float4 per lane = 256 cols/wave
        int h = col >> 6;
        float coef = exw[e * HH + h] / denom[d * HH + h];
        float4 xv = *(const float4*)(xl + (size_t)s * 256 + col);
        float* op = outacc + (size_t)d * 256 + col;
        atomicAdd(op + 0, xv.x * coef);
        atomicAdd(op + 1, xv.y * coef);
        atomicAdd(op + 2, xv.z * coef);
        atomicAdd(op + 3, xv.w * coef);
    }
}

// ---------------- pooling ----------------
__global__ void gat_pool_init_kernel(unsigned* __restrict__ gmax,
                                     float* __restrict__ gsum,
                                     float* __restrict__ cnt) {
    int i = blockIdx.x * blockDim.x + threadIdx.x;
    if (i < GG * FF) { gmax[i] = ORD_NEG_INF; gsum[i] = 0.f; }
    if (i < GG) cnt[i] = 0.f;
}

__global__ void gat_pool_kernel(const float* __restrict__ x,
                                const int* __restrict__ batch,
                                unsigned* __restrict__ gmax,
                                float* __restrict__ gsum,
                                float* __restrict__ cnt) {
    int i = blockIdx.x * blockDim.x + threadIdx.x;   // over N*F
    if (i >= NN * FF) return;
    int n = i >> 6, f = i & 63;
    int g = batch[n];
    float v = x[i];
    atomicMax(&gmax[g * FF + f], f2ord(v));
    atomicAdd(&gsum[g * FF + f], v);
    if (f == 0) atomicAdd(&cnt[g], 1.f);
}

// ---------------- final MLP (one block per graph) ----------------
__global__ void gat_mlp_kernel(const unsigned* __restrict__ gmax,
                               const float* __restrict__ gsum,
                               const float* __restrict__ cnt,
                               const float* __restrict__ W1,
                               const float* __restrict__ b1,
                               const float* __restrict__ W2,
                               const float* __restrict__ b2,
                               float* __restrict__ out) {
    __shared__ float gv[2 * FF];
    __shared__ float hid[DD];
    int g = blockIdx.x, tid = threadIdx.x;
    if (tid < FF) {
        gv[tid] = ord2f(gmax[g * FF + tid]);
    } else if (tid < 2 * FF) {
        float c = cnt[g];
        c = c > 1.f ? c : 1.f;
        gv[tid] = gsum[g * FF + (tid - FF)] / c;
    }
    __syncthreads();
    float a = b1[tid];
    const float* w = W1 + (size_t)tid * (2 * FF);
#pragma unroll 8
    for (int j = 0; j < 2 * FF; ++j) a += gv[j] * w[j];
    hid[tid] = a > 0.f ? a : 0.f;
    __syncthreads();
    if (tid < KK) {
        float o = b2[tid];
        const float* w2 = W2 + (size_t)tid * DD;
        for (int d2 = 0; d2 < DD; ++d2) o += hid[d2] * w2[d2];
        out[g * KK + tid] = o;
    }
}

// ---------------- host driver ----------------
extern "C" void kernel_launch(void* const* d_in, const int* in_sizes, int n_in,
                              void* d_out, int out_size, void* d_ws, size_t ws_size,
                              hipStream_t stream) {
    const float* x      = (const float*)d_in[0];
    const int*   src    = (const int*)d_in[1];            // edge_index[0]
    const int*   dst    = ((const int*)d_in[1]) + EE;     // edge_index[1]
    const int*   batch  = (const int*)d_in[2];
    const float* Wg     = (const float*)d_in[3];          // [L,64,256]
    const float* a_src  = (const float*)d_in[4];          // [L,4,64]
    const float* a_dst  = (const float*)d_in[5];
    const float* bg     = (const float*)d_in[6];          // [L,256]
    const float* Wl     = (const float*)d_in[7];          // [L,64,256]
    const float* bl     = (const float*)d_in[8];          // [L,64]
    const float* W1     = (const float*)d_in[9];          // [256,128]
    const float* b1     = (const float*)d_in[10];
    const float* W2     = (const float*)d_in[11];         // [2,256]
    const float* b2     = (const float*)d_in[12];
    float*       out    = (float*)d_out;

    // workspace carving
    float* ws = (float*)d_ws;
    size_t off = 0;
    float*    xl      = ws + off; off += (size_t)NN * HC;
    float*    outacc  = ws + off; off += (size_t)NN * HC;
    float*    xb1     = ws + off; off += (size_t)NN * FF;
    float*    xb2     = ws + off; off += (size_t)NN * FF;
    float*    als     = ws + off; off += (size_t)NN * HH;
    float*    ald     = ws + off; off += (size_t)NN * HH;
    unsigned* amax    = (unsigned*)(ws + off); off += (size_t)NN * HH;
    float*    denom   = ws + off; off += (size_t)NN * HH;
    float*    alphaw  = ws + off; off += (size_t)EP * HH;
    unsigned* gmaxu   = (unsigned*)(ws + off); off += (size_t)GG * FF;
    float*    gsum    = ws + off; off += (size_t)GG * FF;
    float*    cnt     = ws + off; off += (size_t)GG;

    const float* xin = x;
    float* xouts[2] = { xb1, xb2 };

    for (int l = 0; l < LL; ++l) {
        const float* Wg_l  = Wg    + (size_t)l * FF * HC;
        const float* as_l  = a_src + (size_t)l * HH * CC;
        const float* ad_l  = a_dst + (size_t)l * HH * CC;
        const float* bg_l  = bg    + (size_t)l * HC;
        const float* Wl_l  = Wl    + (size_t)l * FF * HC;
        const float* bl_l  = bl    + (size_t)l * FF;
        float* xout = xouts[l];

        gat_init_layer_kernel<<<(NN * HC + 255) / 256, 256, 0, stream>>>(
            outacc, amax, denom);
        gat_gemm1_kernel<<<NN / 32, 256, G1_LDS, stream>>>(xin, Wg_l, xl);
        gat_alpha_kernel<<<(NN * HH + 255) / 256, 256, 0, stream>>>(
            xl, as_l, ad_l, als, ald);
        gat_edge_max_kernel<<<(EP * HH + 255) / 256, 256, 0, stream>>>(
            src, dst, als, ald, amax, alphaw);
        gat_edge_exp_kernel<<<(EP * HH + 255) / 256, 256, 0, stream>>>(
            dst, amax, alphaw, denom);
        gat_scatter_kernel<<<EP / 8, 256, 0, stream>>>(
            src, dst, xl, alphaw, denom, outacc);
        gat_gemm2_kernel<<<NN / 32, 256, G2_LDS, stream>>>(
            outacc, bg_l, Wl_l, bl_l, xout);
        xin = xout;
    }

    gat_pool_init_kernel<<<(GG * FF + 255) / 256, 256, 0, stream>>>(
        gmaxu, gsum, cnt);
    gat_pool_kernel<<<(NN * FF + 255) / 256, 256, 0, stream>>>(
        xin, batch, gmaxu, gsum, cnt);
    gat_mlp_kernel<<<GG, DD, 0, stream>>>(
        gmaxu, gsum, cnt, W1, b1, W2, b2, out);
}
